// MaskedRQSplineTransform_71270687309941
// MI455X (gfx1250) — matrix operand order repaired and checked by reading, not compiled, not run on hardware
//
#include <hip/hip_runtime.h>
#include <hip/hip_bf16.h>

// ---------------- problem constants ----------------
#define NROWS 262144
#define DDIM  128
#define HDIM  128
#define KBINS 10

typedef __attribute__((ext_vector_type(16))) _Float16 v16h;
typedef __attribute__((ext_vector_type(4)))  _Float16 v4h;
typedef __attribute__((ext_vector_type(8)))  float    v8f;

// ---------------- workspace layout (per MLP) ----------------
#define OFF_W1T 0
#define OFF_W2T 32768
#define OFF_W3T 65536
#define OFF_B1  69632
#define OFF_B2  70144
#define OFF_B3  70656
#define MLP_STRIDE 70720

// ---------------- prep: transpose + f16-convert weights ----------------
__global__ __launch_bounds__(256)
void prep_weights(const float* __restrict__ w1, const float* __restrict__ b1,
                  const float* __restrict__ w2, const float* __restrict__ b2,
                  const float* __restrict__ w3, const float* __restrict__ b3,
                  int kout, char* __restrict__ wsm) {
  _Float16* w1T = (_Float16*)(wsm + OFF_W1T);
  _Float16* w2T = (_Float16*)(wsm + OFF_W2T);
  _Float16* w3T = (_Float16*)(wsm + OFF_W3T);
  float* b1o = (float*)(wsm + OFF_B1);
  float* b2o = (float*)(wsm + OFF_B2);
  float* b3o = (float*)(wsm + OFF_B3);
  int t = blockIdx.x * blockDim.x + threadIdx.x;
  if (t < 16384) {
    int o = t >> 7, i = t & 127;
    w1T[t] = (_Float16)w1[i * HDIM + o];
    w2T[t] = (_Float16)w2[i * HDIM + o];
  }
  if (t < 2048) {
    int o = t >> 7, i = t & 127;
    w3T[t] = (o < kout) ? (_Float16)w3[i * kout + o] : (_Float16)0.f;
  }
  if (t < 128) { b1o[t] = b1[t]; b2o[t] = b2[t]; }
  if (t < 16)  { b3o[t] = (t < kout) ? b3[t] : 0.f; }
}

// ---------------- fused 3xMLP + RQ-spline ----------------
#define LS 136   // padded LDS row stride in halves (272 B -> conflict-free 16-row access)

union Frag { float4 f4[2]; v16h h; };

// One 128->128 layer for a wave's 16-row stripe.
// A-fragments register-resident; B-fragments software-pipelined one N-tile ahead.
__device__ __forceinline__
void layer128(const _Float16* __restrict__ sA, _Float16* __restrict__ sD,
              const _Float16* __restrict__ sW, const float* __restrict__ gB,
              int arow, int n15, int khalf, int r0, int m0) {
  Frag A[4];
  #pragma unroll
  for (int ks = 0; ks < 4; ++ks) {
    const int kb = ks * 32 + khalf * 8;
    A[ks].f4[0] = *(const float4*)&sA[arow * LS + kb];
    A[ks].f4[1] = *(const float4*)&sA[arow * LS + kb + 16];
  }
  Frag B[4];
  #pragma unroll
  for (int ks = 0; ks < 4; ++ks) {
    const int wk = ks * 32 + khalf * 8;
    B[ks].f4[0] = *(const float4*)&sW[n15 * LS + wk];
    B[ks].f4[1] = *(const float4*)&sW[n15 * LS + wk + 16];
  }
  #pragma unroll
  for (int nt = 0; nt < 8; ++nt) {
    Frag Bn[4];
    if (nt < 7) {  // prefetch next tile's B while this tile's WMMAs run
      const int coln = (nt + 1) * 16 + n15;
      #pragma unroll
      for (int ks = 0; ks < 4; ++ks) {
        const int wk = ks * 32 + khalf * 8;
        Bn[ks].f4[0] = *(const float4*)&sW[coln * LS + wk];
        Bn[ks].f4[1] = *(const float4*)&sW[coln * LS + wk + 16];
      }
    }
    v8f acc = {};
    #pragma unroll
    for (int ks = 0; ks < 4; ++ks)
      acc = __builtin_amdgcn_wmma_f32_16x16x32_f16(false, A[ks].h, false, B[ks].h,
                                                   (short)0, acc, false, false);
    const int col = nt * 16 + n15;
    const float bias = gB[col];
    #pragma unroll
    for (int j = 0; j < 8; ++j) {
      float v = fmaxf(acc[j] + bias, 0.f);
      sD[(r0 + m0 + j) * LS + col] = (_Float16)v;
    }
    if (nt < 7) {
      #pragma unroll
      for (int ks = 0; ks < 4; ++ks) B[ks] = Bn[ks];
    }
  }
}

__global__ __launch_bounds__(256)
void fused_mlp_spline(const float* __restrict__ xg, const char* __restrict__ ws,
                      float* __restrict__ outg) {
  __shared__ _Float16 s_x[128 * LS];      // x tile / layer-2 output (ping-pong)
  __shared__ _Float16 s_h[128 * LS];      // layer-1 output
  __shared__ _Float16 s_w[128 * LS];      // staged 128x128 weight (reused per layer)
  __shared__ _Float16 s_w3[16 * LS];      // staged 16x128 layer-3 weight
  __shared__ float    s_l3[8 * 16 * 16];  // per-wave 16x16 layer-3 f32 result
  __shared__ float    s_cw[128 * 12];     // cum widths  [row][0..10]
  __shared__ float    s_hs[128 * 12];     // softmax heights [row][0..9]
  __shared__ float    s_ds[128 * 12];     // softplus derivs [row][0..10]

  const int tid  = threadIdx.x;
  const int lane = tid & 31;
  const int wave = tid >> 5;
  const int gr0  = blockIdx.x * 128;
  const int r0   = wave * 16;

  const int n15   = lane & 15;
  const int khalf = lane >> 4;
  const int m0    = khalf * 8;
  const int arow  = r0 + n15;

  for (int m = 0; m < 3; ++m) {
    const char* wb = ws + m * MLP_STRIDE;
    const _Float16* gW1 = (const _Float16*)(wb + OFF_W1T);
    const _Float16* gW2 = (const _Float16*)(wb + OFF_W2T);
    const _Float16* gW3 = (const _Float16*)(wb + OFF_W3T);
    const float* gB1 = (const float*)(wb + OFF_B1);
    const float* gB2 = (const float*)(wb + OFF_B2);
    const float* gB3 = (const float*)(wb + OFF_B3);

    __syncthreads();
    // ---- stage x tile (f32 -> f16) and W1 into LDS
    for (int idx = tid; idx < 4096; idx += 256) {
      int r = idx >> 5, c4 = idx & 31;
      float4 v = *(const float4*)(xg + (size_t)(gr0 + r) * DDIM + c4 * 4);
      v4h hv = { (_Float16)v.x, (_Float16)v.y, (_Float16)v.z, (_Float16)v.w };
      *(v4h*)&s_x[r * LS + c4 * 4] = hv;
    }
    for (int idx = tid; idx < 2048; idx += 256) {
      int r = idx >> 4, c8 = idx & 15;
      *(float4*)&s_w[r * LS + c8 * 8] = *(const float4*)(gW1 + r * 128 + c8 * 8);
    }
    __syncthreads();

    __builtin_prefetch(gW2 + tid * 64, 0, 0);  // pull W2 toward caches during layer 1

    // ---- layer 1: relu(x @ W1 + b1) -> s_h
    layer128(s_x, s_h, s_w, gB1, arow, n15, khalf, r0, m0);

    __syncthreads();
    // ---- stage W2 and W3
    for (int idx = tid; idx < 2048; idx += 256) {
      int r = idx >> 4, c8 = idx & 15;
      *(float4*)&s_w[r * LS + c8 * 8] = *(const float4*)(gW2 + r * 128 + c8 * 8);
    }
    {
      int r = tid >> 4, c8 = tid & 15;  // exactly 256 chunks for the 16x128 W3
      *(float4*)&s_w3[r * LS + c8 * 8] = *(const float4*)(gW3 + r * 128 + c8 * 8);
    }
    __syncthreads();

    // ---- layer 2: relu(h @ W2 + b2) -> s_x
    layer128(s_h, s_x, s_w, gB2, arow, n15, khalf, r0, m0);

    // ---- layer 3: (h2 @ W3 + b3), single padded 16-wide N tile -> s_l3
    {
      Frag A[4], B[4];
      #pragma unroll
      for (int ks = 0; ks < 4; ++ks) {
        const int kb = ks * 32 + khalf * 8;
        A[ks].f4[0] = *(const float4*)&s_x[arow * LS + kb];
        A[ks].f4[1] = *(const float4*)&s_x[arow * LS + kb + 16];
        const int wk = ks * 32 + khalf * 8;
        B[ks].f4[0] = *(const float4*)&s_w3[n15 * LS + wk];
        B[ks].f4[1] = *(const float4*)&s_w3[n15 * LS + wk + 16];
      }
      v8f acc = {};
      #pragma unroll
      for (int ks = 0; ks < 4; ++ks)
        acc = __builtin_amdgcn_wmma_f32_16x16x32_f16(false, A[ks].h, false, B[ks].h,
                                                     (short)0, acc, false, false);
      const float bias = gB3[n15];
      #pragma unroll
      for (int j = 0; j < 8; ++j)
        s_l3[wave * 256 + (m0 + j) * 16 + n15] = acc[j] + bias;
    }

    // ---- per-row head: softmax / cumsum / softplus into spline tables
    if (lane < 16) {
      const int lr = r0 + lane;
      const float* L = &s_l3[wave * 256 + lane * 16];
      if (m == 2) {  // derivatives: softplus over K+1 = 11 logits
        #pragma unroll
        for (int j = 0; j < KBINS + 1; ++j) {
          float z = L[j];
          s_ds[lr * 12 + j] = (z > 20.f) ? z : log1pf(__expf(z));
        }
      } else {       // widths / heights: stable softmax over K = 10 logits
        float v[KBINS];
        float mx = -3.0e38f;
        #pragma unroll
        for (int j = 0; j < KBINS; ++j) { v[j] = L[j]; mx = fmaxf(mx, v[j]); }
        float s = 0.f;
        #pragma unroll
        for (int j = 0; j < KBINS; ++j) { v[j] = __expf(v[j] - mx); s += v[j]; }
        const float inv = 1.f / s;
        if (m == 0) {
          float c = 0.f;
          s_cw[lr * 12 + 0] = 0.f;
          #pragma unroll
          for (int j = 0; j < KBINS; ++j) { c += v[j] * inv; s_cw[lr * 12 + j + 1] = c; }
        } else {
          #pragma unroll
          for (int j = 0; j < KBINS; ++j) s_hs[lr * 12 + j] = v[j] * inv;
        }
      }
    }
  }

  __syncthreads();
  // ---- rational-quadratic spline, fully elementwise (f32 VALU)
  for (int e = tid; e < 128 * 128; e += 256) {
    const int lr = e >> 7, c = e & 127;
    const float xv = xg[(size_t)(gr0 + lr) * DDIM + c];
    const float xc = fminf(fmaxf(xv, 1e-5f), 1.f - 1e-5f);
    const float* cw = &s_cw[lr * 12];
    int bi = 0;
    #pragma unroll
    for (int j = 1; j <= KBINS; ++j) bi += (xc > cw[j]) ? 1 : 0;
    bi = min(bi, KBINS - 1);
    const float left  = cw[bi];
    const float right = cw[bi + 1];
    const float w  = fmaxf(right - left, 1e-5f);
    const float th = (xc - left) / w;
    const float hk  = s_hs[lr * 12 + bi];
    const float hk1 = s_hs[lr * 12 + min(bi + 1, KBINS - 1)];
    const float dk  = s_ds[lr * 12 + bi];
    const float dk1 = s_ds[lr * 12 + bi + 1];
    const float om  = 1.f - th;
    const float num = hk * th * th + dk * th * om;
    const float den = hk1 * om * om + dk1 * th * om;
    __builtin_nontemporal_store(left + w * num / den,
                                &outg[(size_t)(gr0 + lr) * DDIM + c]);
  }
}

// ---------------- launcher ----------------
extern "C" void kernel_launch(void* const* d_in, const int* in_sizes, int n_in,
                              void* d_out, int out_size, void* d_ws, size_t ws_size,
                              hipStream_t stream) {
  (void)in_sizes; (void)n_in; (void)out_size; (void)ws_size;
  const float* x   = (const float*)d_in[0];
  const float* ww1 = (const float*)d_in[1];  const float* wb1 = (const float*)d_in[2];
  const float* ww2 = (const float*)d_in[3];  const float* wb2 = (const float*)d_in[4];
  const float* ww3 = (const float*)d_in[5];  const float* wb3 = (const float*)d_in[6];
  const float* hw1 = (const float*)d_in[7];  const float* hb1 = (const float*)d_in[8];
  const float* hw2 = (const float*)d_in[9];  const float* hb2 = (const float*)d_in[10];
  const float* hw3 = (const float*)d_in[11]; const float* hb3 = (const float*)d_in[12];
  const float* dw1 = (const float*)d_in[13]; const float* db1 = (const float*)d_in[14];
  const float* dw2 = (const float*)d_in[15]; const float* db2 = (const float*)d_in[16];
  const float* dw3 = (const float*)d_in[17]; const float* db3 = (const float*)d_in[18];

  char* ws = (char*)d_ws;
  prep_weights<<<64, 256, 0, stream>>>(ww1, wb1, ww2, wb2, ww3, wb3, KBINS,     ws + 0 * MLP_STRIDE);
  prep_weights<<<64, 256, 0, stream>>>(hw1, hb1, hw2, hb2, hw3, hb3, KBINS,     ws + 1 * MLP_STRIDE);
  prep_weights<<<64, 256, 0, stream>>>(dw1, db1, dw2, db2, dw3, db3, KBINS + 1, ws + 2 * MLP_STRIDE);

  fused_mlp_spline<<<NROWS / 128, 256, 0, stream>>>(x, ws, (float*)d_out);
}
